// dec_deeplabv3_contrast_dc_85005992722470
// MI455X (gfx1250) — matrix operand order repaired and checked by reading, not compiled, not run on hardware
//
#include <hip/hip_runtime.h>
#include <hip/hip_bf16.h>

// ---------------------------------------------------------------------------
// Problem constants (from the reference)
// ---------------------------------------------------------------------------
#define NC     19            // NUM_CLASSES
#define PD     128           // PROJ_DIM (= K of the GEMM)
#define QL     2975          // QUEUE_LEN
#define NB     8             // batch
#define HW     16384         // 128*128
#define MROWS  (NB * NC)     // 152 key rows (all batch steps at once)
#define MPAD   160           // rows padded to 10 WMMA tiles
#define NTILE  32            // queue columns per block
#define TPC    ((QL + NTILE - 1) / NTILE)  // 93 strips per class
#define NBLK   (NC * TPC)                  // 1767 blocks per projector
#define KST    136           // keys LDS row stride (bf16) - pad vs bank conflicts
#define QST    136           // queue LDS col stride (bf16)
#define SPLIT  16            // blocks per (pi,b) for the proj reduction pass

typedef __attribute__((ext_vector_type(16))) __bf16        v16bf;
typedef __attribute__((ext_vector_type(8)))  float         v8f;
typedef __attribute__((ext_vector_type(4)))  unsigned int  ui4;
typedef __attribute__((ext_vector_type(8)))  int           i8v;
typedef __attribute__((ext_vector_type(4)))  int           i4v;

union Frag {            // bf16 WMMA operand: 16 bf16 = 32B = two 16B LDS loads
  uint4 q[2];
  v16bf b;
};

static __device__ __forceinline__ unsigned short f32_to_bf16_bits(float f) {
  unsigned int u = __float_as_uint(f);
  u += 0x7fffu + ((u >> 16) & 1u);       // round-to-nearest-even
  return (unsigned short)(u >> 16);
}
static __device__ __forceinline__ float bf16_bits_to_f32(unsigned short h) {
  return __uint_as_float(((unsigned int)h) << 16);
}

// ---------------------------------------------------------------------------
// K1: pred = argmax_c logits  (first max wins, like jnp.argmax)
// ---------------------------------------------------------------------------
__global__ __launch_bounds__(256) void argmax_kernel(
    const float* __restrict__ logits, int* __restrict__ pred) {
  int i = blockIdx.x * 256 + threadIdx.x;
  if (i >= NB * HW) return;
  int b = i / HW, h = i - b * HW;
  const float* lp = logits + (size_t)b * NC * HW + h;
  float best = lp[0];
  int bi = 0;
  for (int c = 1; c < NC; ++c) {
    float v = lp[(size_t)c * HW];
    if (v > best) { best = v; bi = c; }
  }
  pred[i] = bi;
}

// ---------------------------------------------------------------------------
// K2: per-(pi,b) per-class feature sums.  Dominant HBM pass (192 MB of proj):
// split each (pi,b) over SPLIT blocks (384 blocks) so the memory system stays
// busy; class-binned accumulation in LDS.
// ---------------------------------------------------------------------------
__global__ __launch_bounds__(256) void partial_sums_kernel(
    const float* __restrict__ p3, const float* __restrict__ p4,
    const float* __restrict__ p5, const int* __restrict__ pred,
    float* __restrict__ gsums, int* __restrict__ gcnts) {
  __shared__ float sums[NC * 132];   // stride 132: bank = (4c + d) % 64
  __shared__ int   cnts[NC];
  int tid = threadIdx.x;
  int s  = blockIdx.x % SPLIT;
  int pb = blockIdx.x / SPLIT;       // pi*8 + b
  int pi = pb / NB, b = pb % NB;
  const float* fea = (pi == 0 ? p3 : (pi == 1 ? p4 : p5)) + (size_t)b * PD * HW;

  for (int i = tid; i < NC * 132; i += 256) sums[i] = 0.f;
  if (tid < NC) cnts[tid] = 0;
  __syncthreads();

  const int chunk = HW / SPLIT;      // 1024 pixels
  int h0 = s * chunk;
  for (int h = h0 + tid; h < h0 + chunk; h += 256) {
    int p = pred[b * HW + h];
    atomicAdd(&cnts[p], 1);
    const float* fcol = fea + h;     // lanes -> consecutive h: coalesced per d
    float* srow = &sums[p * 132];
    for (int d = 0; d < PD; ++d) atomicAdd(&srow[d], fcol[(size_t)d * HW]);
  }
  __syncthreads();

  for (int i = tid; i < NC * PD; i += 256)
    gsums[((size_t)pb * SPLIT + s) * (NC * PD) + i] = sums[(i >> 7) * 132 + (i & 127)];
  if (tid < NC) gcnts[(pb * SPLIT + s) * NC + tid] = cnts[tid];
}

// ---------------------------------------------------------------------------
// K3: reduce partial sums -> masked means -> L2-normalized keys (bf16 bits),
// plus the 'present' table (same for all pi; written by pi==0 blocks).
// ---------------------------------------------------------------------------
__global__ __launch_bounds__(256) void keys_finalize_kernel(
    const float* __restrict__ gsums, const int* __restrict__ gcnts,
    unsigned short* __restrict__ keysB, int* __restrict__ present) {
  __shared__ float mrow[NC * PD];
  __shared__ float cntf[NC];
  __shared__ float nsq[NC];
  int tid = threadIdx.x;
  int pb  = blockIdx.x;              // 24 blocks
  int pi  = pb / NB, b = pb % NB;

  if (tid < NC) {
    int cs = 0;
    for (int s = 0; s < SPLIT; ++s) cs += gcnts[(pb * SPLIT + s) * NC + tid];
    cntf[tid] = (float)cs;
    if (pi == 0) present[b * NC + tid] = (cs > 0) ? 1 : 0;
  }
  __syncthreads();
  for (int i = tid; i < NC * PD; i += 256) {
    float t = 0.f;
    for (int s = 0; s < SPLIT; ++s) t += gsums[((size_t)pb * SPLIT + s) * (NC * PD) + i];
    mrow[i] = t / fmaxf(cntf[i >> 7], 1.f);
  }
  __syncthreads();
  if (tid < NC) {                    // deterministic serial norm per class
    float q = 0.f;
    const float* m = &mrow[tid * PD];
    for (int d = 0; d < PD; ++d) q += m[d] * m[d];
    nsq[tid] = q;
  }
  __syncthreads();
  for (int i = tid; i < NC * PD; i += 256) {
    float key = mrow[i] / fmaxf(sqrtf(nsq[i >> 7]), 1e-12f);
    keysB[(size_t)pb * NC * PD + i] = f32_to_bf16_bits(key);
  }
}

// ---------------------------------------------------------------------------
// K4: GEMM + fused exp-reduction.  M = 152 (all b,c), N = 19*2975, K = 128.
// Tiling is per-class: block = (class k, 32-column strip of its 2975-long
// queue), so the class is uniform per block (no per-lane divide) and the
// strip is one strided 2D tile the Tensor Data Mover can fetch:
//   TDM: data_size=4B, tile 32(l, contiguous) x 128(d, stride QL), OOB
//   columns zero-filled against tensor_dim0 = QL - lbase.
// TDM lands f32 [d][col] in LDS; a transpose+cvt pass produces bf16 [col][d]
// so the WMMA inner loop keeps its two ds_load_b128 per fragment.
// ---------------------------------------------------------------------------
__global__ __launch_bounds__(256) void sim_kernel(
    const unsigned short* __restrict__ keysB,  // [3][152][128] bf16 bits
    const float* __restrict__ queues,          // [3][19][128][2975] f32
    float* __restrict__ diagbuf,               // [3][152][2975]
    float* __restrict__ partAll,               // [3][NBLK][152]
    float* __restrict__ partDiag) {
  __shared__ __align__(16) unsigned short keysLds[MPAD * KST];
  __shared__ __align__(16) unsigned short queueLds[NTILE * QST];
  __shared__ __align__(16) float qstage[PD * NTILE];   // TDM dest: [d][col] f32
  __shared__ float waveAll[8 * MPAD];
  __shared__ float waveDiag[8 * MPAD];

  const int tid   = threadIdx.x;
  const int pi    = blockIdx.y;
  const int kcls  = blockIdx.x / TPC;
  const int lbase = (blockIdx.x % TPC) * NTILE;

  // --- issue the tensor DMA from wave 0 only (TENSORcnt is per-wave) ------
  if (tid < 32) {
    unsigned lds_off = (unsigned)(size_t)(&qstage[0]);  // flat LDS addr low 32 = offset
    unsigned long long ga = (unsigned long long)(const void*)(
        queues + ((size_t)(pi * NC + kcls) * PD) * QL + lbase);
    unsigned td0 = (unsigned)(QL - lbase);   // remaining row len -> OOB zero-fill
    ui4 g0 = { 1u,                                        // count=1, user mode
               lds_off,                                   // lds_addr
               (unsigned)(ga & 0xffffffffu),              // global_addr[31:0]
               (unsigned)((ga >> 32) & 0x01ffffffu) | (2u << 30) };  // addr hi | type=2
    i8v g1 = { (int)0x00020000u,                          // data_size=2 (4 bytes)
               (int)((td0 & 0xffffu) << 16),              // tensor_dim0[15:0]
               (int)(((td0 >> 16) & 0xffffu) | ((unsigned)PD << 16)), // td0 hi | td1 lo
               (int)((unsigned)NTILE << 16),              // td1 hi (0) | tile_dim0=32
               (int)PD,                                   // tile_dim1=128, tile_dim2=0
               (int)QL,                                   // tensor_dim0_stride lo
               0, 0 };                                    // stride hi / dim1_stride
    i4v gz = { 0, 0, 0, 0 };
#if defined(__clang_major__) && (__clang_major__ >= 23)
    i8v gz8 = { 0, 0, 0, 0, 0, 0, 0, 0 };
    __builtin_amdgcn_tensor_load_to_lds(g0, g1, gz, gz, gz8, 0);
#else
    __builtin_amdgcn_tensor_load_to_lds(g0, g1, gz, gz, 0);
#endif
  }

  for (int i = tid; i < 8 * MPAD; i += 256) { waveAll[i] = 0.f; waveDiag[i] = 0.f; }
  for (int i = tid; i < MPAD * PD; i += 256) {
    int r = i >> 7, k = i & 127;
    keysLds[r * KST + k] = (r < MROWS) ? keysB[(size_t)(pi * MROWS + r) * PD + k]
                                       : (unsigned short)0;
  }

  if (tid < 32) __builtin_amdgcn_s_wait_tensorcnt(0);
  __syncthreads();

  // transpose + f32->bf16: qstage[d][col] -> queueLds[col][d]
  for (int i = tid; i < PD * NTILE; i += 256) {
    int d = i >> 5, col = i & 31;
    queueLds[col * QST + d] = f32_to_bf16_bits(qstage[i]);
  }
  __syncthreads();

  const int wave = tid >> 5;
  const int lane = tid & 31;
  const int half = lane >> 4;
  const int l15  = lane & 15;

  for (int t = wave; t < 10 * (NTILE / 16); t += 8) {     // 20 output tiles
    int mt = t >> 1, nt = t & 1;
    v8f acc = {0.f, 0.f, 0.f, 0.f, 0.f, 0.f, 0.f, 0.f};
    for (int kc = 0; kc < PD / 32; ++kc) {
      int kbase = kc * 32;
      Frag A, Bq;
      // A 16x32 bf16: lane<16 -> K {0..7,16..23}+kbase ; lane>=16 -> {8..15,24..31}
      const unsigned short* ap = &keysLds[(mt * 16 + l15) * KST + kbase + 8 * half];
      A.q[0] = *reinterpret_cast<const uint4*>(ap);
      A.q[1] = *reinterpret_cast<const uint4*>(ap + 16);
      // B 32x16 bf16: lane<16 -> K {0..7,16..23}+kbase ; lane>=16 -> {8..15,24..31}
      const unsigned short* bp = &queueLds[(nt * 16 + l15) * QST + kbase + 8 * half];
      Bq.q[0] = *reinterpret_cast<const uint4*>(bp);
      Bq.q[1] = *reinterpret_cast<const uint4*>(bp + 16);
      acc = __builtin_amdgcn_wmma_f32_16x16x32_bf16(
          false, A.b, false, Bq.b, (short)0, acc, false, false);
    }
    // Consumer: sim -> exp -> per-row cross-lane reduction (deterministic).
    int l = lbase + nt * 16 + l15;
    bool colValid = (l < QL);
#pragma unroll
    for (int j = 0; j < 8; ++j) {
      int r = mt * 16 + j + 8 * half;      // C layout: VGPR j -> M=j (+8 hi half)
      float e = 0.f, ed = 0.f;
      if (colValid && r < MROWS) {
        float sim = acc[j] * (1.0f / (float)PD);
        e = __expf(sim);
        int b = r / NC, c = r - b * NC;
        if (c == kcls) {                   // class is uniform per block now
          ed = e;
          diagbuf[(size_t)(pi * MROWS + r) * QL + l] = sim;
        }
      }
      for (int off = 1; off < 16; off <<= 1) {   // reduce over the 16 columns
        e  += __shfl_xor(e,  off);
        ed += __shfl_xor(ed, off);
      }
      if (l15 == 0) {                            // lanes 0 and 16: rows j, j+8
        waveAll [wave * MPAD + r] += e;
        waveDiag[wave * MPAD + r] += ed;
      }
    }
  }
  __syncthreads();
  for (int r = tid; r < MROWS; r += 256) {
    float sa = 0.f, sd = 0.f;
    for (int w = 0; w < 8; ++w) { sa += waveAll[w * MPAD + r]; sd += waveDiag[w * MPAD + r]; }
    size_t o = ((size_t)pi * NBLK + blockIdx.x) * MROWS + r;
    partAll[o] = sa;
    partDiag[o] = sd;
  }
}

// ---------------------------------------------------------------------------
// K5: enqueue fixup.  At step b, queue column j (< #present earlier steps) of
// class k holds the key of the j-th earlier present step instead of the
// original column.  Recompute those <= 19*19*7 sims per (pi,b), emit exp
// deltas + overwrite the affected diagonal entries.
// ---------------------------------------------------------------------------
__global__ __launch_bounds__(256) void fixup_kernel(
    const float* __restrict__ queues, const unsigned short* __restrict__ keysB,
    const int* __restrict__ present,
    float* __restrict__ diagbuf, float* __restrict__ corrAll,
    float* __restrict__ corrDiag) {
  __shared__ float cAll[NC], cDiag[NC];
  int tid = threadIdx.x;
  int pi = blockIdx.x / NB, b = blockIdx.x % NB;
  if (tid < NC) { cAll[tid] = 0.f; cDiag[tid] = 0.f; }
  __syncthreads();

  for (int idx = tid; idx < NC * 8 * NC; idx += 256) {
    int k = idx / (8 * NC);
    int j = (idx / NC) & 7;
    int c = idx % NC;
    int cnt = 0, bj = -1;                      // j-th earlier present step for class k
    for (int bp = 0; bp < b; ++bp)
      if (present[bp * NC + k]) { if (cnt == j) bj = bp; ++cnt; }
    if (bj < 0) continue;                      // column j not replaced at step b

    const unsigned short* rowc = keysB + (size_t)((pi * NB + b)  * NC + c) * PD;
    const unsigned short* rowk = keysB + (size_t)((pi * NB + bj) * NC + k) * PD;
    const float* qcol = queues + ((size_t)(pi * NC + k) * PD) * QL + j;
    float dOld = 0.f, dNew = 0.f;
    for (int d = 0; d < PD; ++d) {
      float a = bf16_bits_to_f32(rowc[d]);
      dOld += a * bf16_bits_to_f32(f32_to_bf16_bits(qcol[(size_t)d * QL]));
      dNew += a * bf16_bits_to_f32(rowk[d]);
    }
    float sNew = dNew * (1.f / (float)PD);
    float delta = __expf(sNew) - __expf(dOld * (1.f / (float)PD));
    atomicAdd(&cAll[c], delta);
    if (k == c) {
      atomicAdd(&cDiag[c], delta);
      diagbuf[(size_t)((pi * NB + b) * NC + c) * QL + j] = sNew;
    }
  }
  __syncthreads();
  if (tid < NC) {
    corrAll [(pi * NB + b) * NC + tid] = cAll[tid];
    corrDiag[(pi * NB + b) * NC + tid] = cDiag[tid];
  }
}

// ---------------------------------------------------------------------------
// K6: per-(pi,b,c) loss: reduce the per-strip exp sums, form s_neg, then the
// -(pos - log(exp(pos)+s_neg)) scan over the diagonal row.
// ---------------------------------------------------------------------------
static __device__ __forceinline__ float block_reduce(float v, float* red) {
  int tid = threadIdx.x;
  red[tid] = v;
  __syncthreads();
  for (int s = 128; s > 0; s >>= 1) {
    if (tid < s) red[tid] += red[tid + s];
    __syncthreads();
  }
  float r = red[0];
  __syncthreads();
  return r;
}

__global__ __launch_bounds__(256) void loss_kernel(
    const float* __restrict__ partAll, const float* __restrict__ partDiag,
    const float* __restrict__ corrAll, const float* __restrict__ corrDiag,
    const float* __restrict__ diagbuf, const int* __restrict__ present,
    float* __restrict__ lossbuf) {
  __shared__ float red[256];
  int tid = threadIdx.x;
  int idx = blockIdx.x;            // pi*152 + (b*19 + c), 456 blocks
  int pi = idx / MROWS;
  int r  = idx - pi * MROWS;
  int b = r / NC, c = r - b * NC;

  float sa = 0.f, sd = 0.f;
  for (int blk = tid; blk < NBLK; blk += 256) {
    size_t o = ((size_t)pi * NBLK + blk) * MROWS + r;
    sa += partAll[o];
    sd += partDiag[o];
  }
  float sumAll  = block_reduce(sa, red);
  float sumDiag = block_reduce(sd, red);
  float s_neg = (sumAll + corrAll[idx]) - (sumDiag + corrDiag[idx]);

  const float* dg = diagbuf + (size_t)idx * QL;
  float acc = 0.f;
  for (int l = tid; l < QL; l += 256) {
    float pos = dg[l];
    acc += pos - __logf(__expf(pos) + s_neg);
  }
  float tot = block_reduce(acc, red);
  if (tid == 0)
    lossbuf[idx] = present[b * NC + c] ? (-tot / (float)QL) : 0.f;
}

// K7: deterministic final scalar.
__global__ void total_kernel(const float* __restrict__ lossbuf, float* __restrict__ out) {
  if (threadIdx.x == 0 && blockIdx.x == 0) {
    float t = 0.f;
    for (int i = 0; i < 3 * MROWS; ++i) t += lossbuf[i];
    out[0] = t / (float)NB;
  }
}

// ---------------------------------------------------------------------------
extern "C" void kernel_launch(void* const* d_in, const int* in_sizes, int n_in,
                              void* d_out, int out_size, void* d_ws, size_t ws_size,
                              hipStream_t stream) {
  (void)in_sizes; (void)n_in; (void)out_size; (void)ws_size;
  const float* proj3  = (const float*)d_in[0];
  const float* proj4  = (const float*)d_in[1];
  const float* proj5  = (const float*)d_in[2];
  const float* logits = (const float*)d_in[3];
  const float* queues = (const float*)d_in[4];
  float* out = (float*)d_out;

  char* ws = (char*)d_ws;
  size_t off = 0;
  auto alloc = [&](size_t bytes) -> void* {
    void* p = ws + off;
    off = (off + bytes + 255) & ~(size_t)255;
    return p;
  };
  int*            pred     = (int*)alloc((size_t)NB * HW * 4);
  int*            present  = (int*)alloc((size_t)NB * NC * 4);
  float*          gsums    = (float*)alloc((size_t)24 * SPLIT * NC * PD * 4);
  int*            gcnts    = (int*)alloc((size_t)24 * SPLIT * NC * 4);
  unsigned short* keysB    = (unsigned short*)alloc((size_t)3 * MROWS * PD * 2);
  float*          diagbuf  = (float*)alloc((size_t)3 * MROWS * QL * 4);
  float*          partAll  = (float*)alloc((size_t)3 * NBLK * MROWS * 4);
  float*          partDiag = (float*)alloc((size_t)3 * NBLK * MROWS * 4);
  float*          corrAll  = (float*)alloc((size_t)3 * MROWS * 4);
  float*          corrDiag = (float*)alloc((size_t)3 * MROWS * 4);
  float*          lossbuf  = (float*)alloc((size_t)3 * MROWS * 4);

  argmax_kernel<<<(NB * HW + 255) / 256, 256, 0, stream>>>(logits, pred);
  partial_sums_kernel<<<24 * SPLIT, 256, 0, stream>>>(proj3, proj4, proj5, pred,
                                                      gsums, gcnts);
  keys_finalize_kernel<<<24, 256, 0, stream>>>(gsums, gcnts, keysB, present);
  sim_kernel<<<dim3(NBLK, 3), 256, 0, stream>>>(keysB, queues, diagbuf,
                                                partAll, partDiag);
  fixup_kernel<<<24, 256, 0, stream>>>(queues, keysB, present, diagbuf,
                                       corrAll, corrDiag);
  loss_kernel<<<3 * MROWS, 256, 0, stream>>>(partAll, partDiag, corrAll, corrDiag,
                                             diagbuf, present, lossbuf);
  total_kernel<<<1, 64, 0, stream>>>(lossbuf, out);
}
